// FLGC_13073880449671
// MI455X (gfx1250) — matrix operand order repaired and checked by reading, not compiled, not run on hardware
//
#include <hip/hip_runtime.h>
#include <hip/hip_bf16.h>

// ---------------------------------------------------------------------------
// FLGC forward on gfx1250: 1x1 conv == GEMM (M=512, K=512, N=B*H*W=147456).
// Gates + output permutation folded into a precomputed bf16 weight matrix W2.
// Main loop: V_WMMA_F32_16X16X32_BF16 with double-buffered LDS tiles;
// A tile staged via global_load_async_to_lds_b128 (ASYNCcnt), x tile staged
// f32->bf16 transposed in LDS, non-temporal output stores.
// ---------------------------------------------------------------------------

typedef __attribute__((ext_vector_type(16))) __bf16 v16bf;
typedef __attribute__((ext_vector_type(8)))  float  v8f;

union FragU {
    uint4  q[2];   // 32 bytes
    v16bf  v;
};

#define C_IN   512
#define C_OUT  512
#define NBATCH 16
#define HW     9216      // 96*96
#define KP     40        // LDS K-stride in ushorts (80B rows: 16B-aligned frags)
#define BM     128
#define BN     128
#define BK     32
#define KSTEPS (C_IN / BK)
#define NTILES (HW / BN) // 72 N-tiles per batch

__device__ __forceinline__ unsigned short bfbits(float f) {
    union { __hip_bfloat16 h; unsigned short u; } cv;
    cv.h = __float2bfloat16(f);
    return cv.u;
}

// ---------------------------------------------------------------------------
// Kernel 1: softmax gates, argmax groups, stable argsort p of t, rmap = p[p].
// One block, 512 threads.
// ---------------------------------------------------------------------------
__global__ void flgc_prep(const float* __restrict__ S, const float* __restrict__ T,
                          int* __restrict__ s_idx, float* __restrict__ s_gain,
                          int* __restrict__ rmap, int* __restrict__ t_row,
                          float* __restrict__ tg_row) {
    __shared__ int   t_sh[C_OUT];
    __shared__ float tg_sh[C_OUT];
    __shared__ int   p_sh[C_OUT];
    __shared__ int   offs[17];

    const int c = threadIdx.x;   // 0..511

    {   // softmax row of S -> group id + gate value at argmax
        const float* row = S + c * 16;
        float m = row[0]; int am = 0;
        for (int g = 1; g < 16; ++g) { float v = row[g]; if (v > m) { m = v; am = g; } }
        float sum = 0.f;
        for (int g = 0; g < 16; ++g) sum += __expf(row[g] - m);
        s_idx[c]  = am;
        s_gain[c] = 1.0f / sum;          // exp(m-m)/sum
    }
    {   // softmax row of T
        const float* row = T + c * 16;
        float m = row[0]; int am = 0;
        for (int g = 1; g < 16; ++g) { float v = row[g]; if (v > m) { m = v; am = g; } }
        float sum = 0.f;
        for (int g = 0; g < 16; ++g) sum += __expf(row[g] - m);
        t_sh[c]  = am;
        tg_sh[c] = 1.0f / sum;
    }
    __syncthreads();

    if (threadIdx.x < 16) {              // per-group counts
        int g = threadIdx.x, cnt = 0;
        for (int o = 0; o < C_OUT; ++o) cnt += (t_sh[o] == g);
        offs[g + 1] = cnt;
    }
    __syncthreads();
    if (threadIdx.x == 0) {              // exclusive prefix sum (16 entries)
        offs[0] = 0;
        for (int g = 0; g < 16; ++g) offs[g + 1] += offs[g];
    }
    __syncthreads();
    if (threadIdx.x < 16) {              // stable counting sort -> p
        int g = threadIdx.x, pos = offs[g];
        for (int o = 0; o < C_OUT; ++o) if (t_sh[o] == g) p_sh[pos++] = o;
    }
    __syncthreads();
    {
        int j = threadIdx.x;
        int r = p_sh[p_sh[j]];           // out channel j <- dense filter p[p[j]]
        rmap[j]   = r;
        t_row[j]  = t_sh[r];
        tg_row[j] = tg_sh[r];
    }
}

// ---------------------------------------------------------------------------
// Kernel 2: fused permuted/masked/scaled weight -> bf16 W2[512][512] (512 KB).
// ---------------------------------------------------------------------------
__global__ __launch_bounds__(256)
void flgc_weight(const float* __restrict__ conv,
                 const int* __restrict__ s_idx, const float* __restrict__ s_gain,
                 const int* __restrict__ rmap, const int* __restrict__ t_row,
                 const float* __restrict__ tg_row,
                 unsigned short* __restrict__ W2) {
    int idx = blockIdx.x * 256 + threadIdx.x;   // 0 .. 512*512-1, idx = j*512+c
    int j = idx >> 9;
    int c = idx & 511;
    int r = rmap[j];
    float w = (t_row[j] == s_idx[c]) ? conv[r * C_IN + c] * tg_row[j] * s_gain[c]
                                     : 0.0f;
    W2[idx] = bfbits(w);
}

// ---------------------------------------------------------------------------
// Kernel 3: GEMM. 256 threads = 8 waves (2 waveM x 4 waveN).
// Block tile 128M x 128N, K-step 32. Wave tile 64M x 32N -> 8 WMMA/k-step.
// Double-buffered LDS: stage tile i+1 while computing tile i; one barrier
// and one s_wait_asynccnt per k-step, both after the WMMA block.
// grid = (512/128, B * HW/128) = (4, 1152); M fastest for L2 reuse of x.
// ---------------------------------------------------------------------------
__global__ __launch_bounds__(256)
void flgc_gemm(const float* __restrict__ x, const unsigned short* __restrict__ W2,
               float* __restrict__ out) {
    __shared__ unsigned short sA[2][BM * KP];  // W2 tiles, async-filled
    __shared__ unsigned short sB[2][BN * KP];  // x tiles, bf16 transposed

    const int tid   = threadIdx.x;
    const int lane  = tid & 31;
    const int wid   = tid >> 5;
    const int waveM = wid & 1;                  // 0..1 -> 64-row slice
    const int waveN = wid >> 1;                 // 0..3 -> 32-col slice
    const int half  = lane >> 4;                // ISA lane-half for 16-bit frags
    const int nl    = lane & 15;

    const int M0  = blockIdx.x * BM;
    const int b   = blockIdx.y / NTILES;
    const int hw0 = (blockIdx.y % NTILES) * BN;

    const float* xb = x + (size_t)b * C_IN * HW;

    // staging assignments (fixed per thread)
    const int sn  = tid & 127;                  // x-stage: column n (coalesced)
    const int skb = (tid >> 7) * 16;            // x-stage: K run base (0 or 16)
    const int sm  = tid & 127;                  // A-stage: row m
    const int sc0 = tid >> 7;                   // A-stage: first 16B chunk id

    v8f acc[4][2] = {};

    // ---- staging of one k-tile into buffer `sel` ----------------------------
    auto stage = [&](int k0, int sel) {
        // async A tile: W2[M0..M0+127][k0..k0+31] -> sA[sel] (ASYNCcnt)
        #pragma unroll
        for (int cc = 0; cc < 2; ++cc) {
            int c = sc0 + 2 * cc;                                   // 0..3
            unsigned ldsa = (unsigned)(uintptr_t)&sA[sel][sm * KP + c * 8];
            const unsigned short* g =
                W2 + (size_t)(M0 + sm) * C_IN + k0 + c * 8;
            asm volatile("global_load_async_to_lds_b128 %0, %1, off"
                         :: "v"(ldsa), "v"(g) : "memory");
        }
        // x tile [32K x 128N] f32 -> sB[sel] bf16 transposed.
        // thread owns 16 consecutive K for one column: packs 4x ds_store_b64.
        float xr[16];
        #pragma unroll
        for (int kk = 0; kk < 16; ++kk)
            xr[kk] = xb[(size_t)(k0 + skb + kk) * HW + hw0 + sn];
        #pragma unroll
        for (int i = 0; i < 4; ++i) {
            uint2 pk;
            pk.x = (unsigned)bfbits(xr[4 * i + 0]) |
                   ((unsigned)bfbits(xr[4 * i + 1]) << 16);
            pk.y = (unsigned)bfbits(xr[4 * i + 2]) |
                   ((unsigned)bfbits(xr[4 * i + 3]) << 16);
            *reinterpret_cast<uint2*>(&sB[sel][sn * KP + skb + 4 * i]) = pk;
        }
    };

    // prologue: fill buffer 0
    stage(0, 0);
    asm volatile("s_wait_asynccnt 0x0" ::: "memory");
    __syncthreads();

    for (int it = 0; it < KSTEPS; ++it) {
        const int cur = it & 1;

        // ---- stage next tile into the other buffer (overlaps compute) ------
        if (it + 1 < KSTEPS) stage((it + 1) * BK, cur ^ 1);

        // ---- A fragments from LDS ------------------------------------------
        // A frag in this code's contiguous k order: lane nl = row M; elems 0-15: K=16h..16h+15
        // (matches B) -> two 16B-aligned chunks.
        FragU fa[4];
        #pragma unroll
        for (int mm = 0; mm < 4; ++mm) {
            int ml = waveM * 64 + mm * 16 + nl;
            fa[mm].q[0] = *(const uint4*)&sA[cur][ml * KP + 16 * half];
            fa[mm].q[1] = *(const uint4*)&sA[cur][ml * KP + 16 * half + 8];
        }

        // ---- B fragments from LDS ------------------------------------------
        // B frag in this code's contiguous k order: lane nl = col N; lanes 0-15: K=0..15,
        // lanes 16-31: K=16..31 -> one contiguous 32B read per lane.
        FragU fb[2];
        #pragma unroll
        for (int nn = 0; nn < 2; ++nn) {
            int n_local = waveN * 32 + nn * 16 + nl;
            const uint4* p = (const uint4*)&sB[cur][n_local * KP + 16 * half];
            fb[nn].q[0] = p[0];
            fb[nn].q[1] = p[1];
        }

        // ---- 8x V_WMMA_F32_16X16X32_BF16 -----------------------------------
        #pragma unroll
        for (int mm = 0; mm < 4; ++mm)
            #pragma unroll
            for (int nn = 0; nn < 2; ++nn)
                acc[mm][nn] = __builtin_amdgcn_wmma_f32_16x16x32_bf16(
                    false, fa[mm].v, false, fb[nn].v,
                    (short)0, acc[mm][nn], false, false);

        // ---- close the pipeline stage: async + LDS stores of tile it+1 -----
        if (it + 1 < KSTEPS) {
            asm volatile("s_wait_asynccnt 0x0" ::: "memory");
            __syncthreads();
        }
    }

    // ---- store (non-temporal: out is 302MB write-once, keep x in L2) --------
    // D layout: lane nl = N col, VGPR r -> M = r + 8*half.
    float* ob = out + (size_t)b * C_OUT * HW;
    #pragma unroll
    for (int mm = 0; mm < 4; ++mm) {
        #pragma unroll
        for (int nn = 0; nn < 2; ++nn) {
            int hw = hw0 + waveN * 32 + nn * 16 + nl;
            int jb = M0 + waveM * 64 + mm * 16 + 8 * half;
            #pragma unroll
            for (int r = 0; r < 8; ++r)
                __builtin_nontemporal_store(acc[mm][nn][r],
                                            &ob[(size_t)(jb + r) * HW + hw]);
        }
    }
}

// ---------------------------------------------------------------------------
extern "C" void kernel_launch(void* const* d_in, const int* in_sizes, int n_in,
                              void* d_out, int out_size, void* d_ws, size_t ws_size,
                              hipStream_t stream) {
    const float* x    = (const float*)d_in[0];   // (16,512,96,96)
    const float* conv = (const float*)d_in[1];   // (512,512,1,1)
    const float* S    = (const float*)d_in[2];   // (512,16)
    const float* T    = (const float*)d_in[3];   // (512,16)
    float* out = (float*)d_out;                  // (16,512,96,96)

    // workspace layout
    char* base = (char*)d_ws;
    unsigned short* W2     = (unsigned short*)base;                 // 512KB
    float*          s_gain = (float*)(base + (size_t)C_OUT * C_IN * 2);
    float*          tg_row = s_gain + C_IN;
    int*            s_idx  = (int*)(tg_row + C_OUT);
    int*            t_row  = s_idx + C_IN;
    int*            rmap   = t_row + C_OUT;

    flgc_prep<<<1, 512, 0, stream>>>(S, T, s_idx, s_gain, rmap, t_row, tg_row);

    flgc_weight<<<(C_OUT * C_IN) / 256, 256, 0, stream>>>(
        conv, s_idx, s_gain, rmap, t_row, tg_row, W2);

    dim3 grid(C_OUT / BM, NBATCH * NTILES);      // (4, 1152), M fastest
    flgc_gemm<<<grid, 256, 0, stream>>>(x, W2, out);
}
